// Attention_1649267441685
// MI455X (gfx1250) — matrix operand order repaired and checked by reading, not compiled, not run on hardware
//
#include <hip/hip_runtime.h>

// ---------------------------------------------------------------------------
// Problem constants (from the reference)
// ---------------------------------------------------------------------------
#define B_DIM     8
#define N_LAT     512
#define M_BYTE    4096
#define D_LATENT  1024
#define D_BYTE    512
#define H_DIM     8
#define DK_DIM    128
#define HD_DIM    1024            // H * DK == H * DV
#define TEMP_SC   0.08838834764831845f
#define LOG2E     1.4426950408889634f

typedef __bf16 bf16_t;
typedef __attribute__((ext_vector_type(16))) __bf16 v16bf;
typedef __attribute__((ext_vector_type(8)))  float  v8f;
typedef __attribute__((ext_vector_type(4))) unsigned int u32x4;
typedef __attribute__((ext_vector_type(8))) int i32x8;
typedef __attribute__((ext_vector_type(4))) int i32x4;

union FragU {
  v16bf v;
  bf16_t e[16];
  unsigned int w[8];
};

// ---------------------------------------------------------------------------
// WMMA wrapper: D = A(16x32 bf16) * B(32x16 bf16) + C(16x16 f32)
// ---------------------------------------------------------------------------
__device__ inline v8f wmma_bf16(v16bf a, v16bf b, v8f c) {
  return __builtin_amdgcn_wmma_f32_16x16x32_bf16(
      /*neg_a=*/false, a, /*neg_b=*/false, b,
      /*c_mod=*/(short)0, c, /*reuse_a=*/false, /*reuse_b=*/false);
}

// DPP16 lane permute (VALU, no LDS port pressure). CTRL stays within the
// 16-lane DPP row, so the two wave halves remain independent.
template <int CTRL>
__device__ inline float dpp_shuf(float x) {
  return __builtin_bit_cast(float,
      __builtin_amdgcn_update_dpp(0, __builtin_bit_cast(int, x),
                                  CTRL, 0xF, 0xF, true));
}

// Butterfly reduction over the 16 lanes of each half-wave.
// Masks {1,2,7,15} span the group: quad_perm(1,0,3,2), quad_perm(2,3,0,1),
// row_half_mirror, row_mirror.
__device__ inline float half_reduce_max(float x) {
  x = fmaxf(x, dpp_shuf<0x0B1>(x));   // xor 1
  x = fmaxf(x, dpp_shuf<0x04E>(x));   // xor 2
  x = fmaxf(x, dpp_shuf<0x141>(x));   // xor 7 (row_half_mirror)
  x = fmaxf(x, dpp_shuf<0x140>(x));   // xor 15 (row_mirror)
  return x;
}
__device__ inline float half_reduce_sum(float x) {
  x += dpp_shuf<0x0B1>(x);
  x += dpp_shuf<0x04E>(x);
  x += dpp_shuf<0x141>(x);
  x += dpp_shuf<0x140>(x);
  return x;
}

// A fragment (16x32, row-major LDS tile, pitch in elements, pitch even).
// Lane L: M = L&15, half = L>>4; VGPR j holds K = half*16 + 2j (+1)
__device__ inline v16bf load_a_frag(const bf16_t* lds, int pitch, int row0, int col0, int lane) {
  int m = lane & 15, hf = lane >> 4;
  FragU f;
#pragma unroll
  for (int j = 0; j < 8; ++j) {
    int k = hf * 16 + (j << 1);
    f.w[j] = *(const unsigned int*)(lds + (size_t)(row0 + m) * pitch + col0 + k);
  }
  return f.v;
}

// B fragment from LDS tile laid out [K][N] (pitch in elements).
// Lane L: N = L&15, half = L>>4; VGPR j holds K = half*16 + 2j, 2j+1
__device__ inline v16bf load_b_frag(const bf16_t* lds, int pitch, int row0, int col0, int lane) {
  int n = lane & 15, hf = lane >> 4;
  FragU f;
#pragma unroll
  for (int j = 0; j < 8; ++j) {
    int k = hf * 16 + (j << 1);
    f.e[2 * j]     = lds[(size_t)(row0 + k)     * pitch + col0 + n];
    f.e[2 * j + 1] = lds[(size_t)(row0 + k + 1) * pitch + col0 + n];
  }
  return f.v;
}

// B fragment when the LDS tile is laid out [N][K] (e.g. K-cache rows = keys,
// cols = d) -> the two packed K values are contiguous, so use dword loads.
__device__ inline v16bf load_bT_frag(const bf16_t* lds, int pitch, int nrow0, int kcol0, int lane) {
  int n = lane & 15, hf = lane >> 4;
  FragU f;
#pragma unroll
  for (int j = 0; j < 8; ++j) {
    int k = hf * 16 + (j << 1);
    f.w[j] = *(const unsigned int*)(lds + (size_t)(nrow0 + n) * pitch + kcol0 + k);
  }
  return f.v;
}

// ---------------------------------------------------------------------------
// Tensor Data Mover: 2D tile load Global -> LDS (D# groups per ISA 08_async).
//   LDS padding: 64 dwords (256B) data then 4 dwords (16B) pad -> 272B rows,
//   matching the 136-element LDS pitch of the consumer tiles.
//   6-arg builtin form: (u32x4 g0, i32x8 g1, i32x4 g2, i32x4 g3, i32x8, i32 cpol)
// ---------------------------------------------------------------------------
__device__ inline void tdm_load_tile_2d(unsigned lds_off, const bf16_t* gaddr,
                                        unsigned tensor_w, unsigned tensor_h,
                                        unsigned tile_w, unsigned tile_h,
                                        unsigned long long row_stride_elems) {
  unsigned long long ga = (unsigned long long)(const void*)gaddr;
  u32x4 g0;
  g0[0] = 1u;                                          // count=1, user mode
  g0[1] = lds_off;                                     // lds_addr (bytes)
  g0[2] = (unsigned)(ga & 0xffffffffu);                // global_addr[31:0]
  g0[3] = (unsigned)((ga >> 32) & 0x01ffffffu)         // global_addr[56:32]
        | (2u << 30);                                  // type = 2 ("image")
  i32x8 g1;
  g1[0] = (1 << 16)                                    // data_size = 2 bytes
        | (1 << 20)                                    // pad_enable
        | (5 << 22)                                    // pad_interval: 64 dwords
        | (3 << 25);                                   // pad_amount: 4 dwords
  g1[1] = (int)((tensor_w & 0xffffu) << 16);           // tensor_dim0[15:0]
  g1[2] = (int)(((tensor_w >> 16) & 0xffffu)           // tensor_dim0[31:16]
        | ((tensor_h & 0xffffu) << 16));               // tensor_dim1[15:0]
  g1[3] = (int)(((tensor_h >> 16) & 0xffffu)           // tensor_dim1[31:16]
        | ((tile_w & 0xffffu) << 16));                 // tile_dim0
  g1[4] = (int)(tile_h & 0xffffu);                     // tile_dim1 (tile_dim2=0)
  g1[5] = (int)(row_stride_elems & 0xffffffffu);       // tensor_dim0_stride lo
  g1[6] = (int)((row_stride_elems >> 32) & 0xffffu);   // stride hi; dim1_stride=0
  g1[7] = 0;
  i32x4 z4 = {0, 0, 0, 0};                             // groups 2/3: 2D tensor
  i32x8 z8 = {0, 0, 0, 0, 0, 0, 0, 0};
  __builtin_amdgcn_tensor_load_to_lds(g0, g1, z4, z4, z8, /*cpol=*/0);
}

// ---------------------------------------------------------------------------
// fp32 -> bf16 conversion (exact grids; all sizes divisible by 1024)
// ---------------------------------------------------------------------------
__global__ __launch_bounds__(256) void f32_to_bf16_kernel(
    const float* __restrict__ in, bf16_t* __restrict__ out) {
  size_t i = ((size_t)blockIdx.x * 256 + threadIdx.x) * 4;
  float4 f = *(const float4*)(in + i);
  out[i + 0] = (bf16_t)f.x;
  out[i + 1] = (bf16_t)f.y;
  out[i + 2] = (bf16_t)f.z;
  out[i + 3] = (bf16_t)f.w;
}

// ---------------------------------------------------------------------------
// Generic bf16 GEMM: C[M,N] = A[M,K] (row-major) * B[K,N] (row-major)
// 256 threads = 8 waves (wave32), 128x128 block tile, 32-wide K steps.
// Wave grid 4x2: each wave computes a 32x64 sub-tile (2x4 WMMA frags).
// OUT_BF16: store bf16. Else: f32 store with residual add.
// ---------------------------------------------------------------------------
#define GP_A 40      // A tile pitch (elems); 80B rows, 16B-aligned
#define GP_B 136     // B tile pitch (elems); 272B rows, 16B-aligned

template <bool OUT_BF16>
__global__ __launch_bounds__(256) void gemm_bf16_kernel(
    const bf16_t* __restrict__ A, const bf16_t* __restrict__ Bm,
    const float* __restrict__ resid, void* __restrict__ Out,
    int M, int N, int K) {
  __shared__ __align__(16) bf16_t Alds[128][GP_A];
  __shared__ __align__(16) bf16_t Blds[32][GP_B];

  int t = threadIdx.x;
  int lane = t & 31, wave = t >> 5;
  int wm = wave >> 1, wn = wave & 1;
  int m0 = blockIdx.y * 128, n0 = blockIdx.x * 128;

  v8f acc[2][4] = {};

  for (int k0 = 0; k0 < K; k0 += 32) {
    __syncthreads();
    // Stage A tile 128x32 (512 x uint4, 2 per thread)
#pragma unroll
    for (int i = 0; i < 2; ++i) {
      int idx = t + i * 256;
      int r = idx >> 2, c = (idx & 3) << 3;
      *(uint4*)(&Alds[r][c]) = *(const uint4*)(A + (size_t)(m0 + r) * K + k0 + c);
    }
    // Stage B tile 32x128 (512 x uint4, 2 per thread)
#pragma unroll
    for (int i = 0; i < 2; ++i) {
      int idx = t + i * 256;
      int r = idx >> 4, c = (idx & 15) << 3;
      *(uint4*)(&Blds[r][c]) = *(const uint4*)(Bm + (size_t)(k0 + r) * N + n0 + c);
    }
    // Prefetch next k-step tiles into cache (global_prefetch_b8)
    if (k0 + 32 < K) {
      __builtin_prefetch(A + (size_t)(m0 + (t >> 1)) * K + k0 + 32 + ((t & 1) << 4), 0, 1);
      __builtin_prefetch(Bm + (size_t)(k0 + 32 + (t >> 4)) * N + n0 + ((t & 15) << 3), 0, 1);
    }
    __syncthreads();

    v16bf a0 = load_a_frag(&Alds[0][0], GP_A, wm * 32, 0, lane);
    v16bf a1 = load_a_frag(&Alds[0][0], GP_A, wm * 32 + 16, 0, lane);
#pragma unroll
    for (int j2 = 0; j2 < 4; ++j2) {
      v16bf b = load_b_frag(&Blds[0][0], GP_B, 0, wn * 64 + j2 * 16, lane);
      acc[0][j2] = wmma_bf16(a0, b, acc[0][j2]);
      acc[1][j2] = wmma_bf16(a1, b, acc[1][j2]);
    }
  }

  // Epilogue: C/D layout -> row = i + 8*half, col = lane&15
  int hf = lane >> 4, nn = lane & 15;
#pragma unroll
  for (int i2 = 0; i2 < 2; ++i2) {
#pragma unroll
    for (int j2 = 0; j2 < 4; ++j2) {
#pragma unroll
      for (int i = 0; i < 8; ++i) {
        int row = m0 + wm * 32 + i2 * 16 + i + 8 * hf;
        int col = n0 + wn * 64 + j2 * 16 + nn;
        float val = acc[i2][j2][i];
        size_t o = (size_t)row * N + col;
        if (OUT_BF16) {
          ((bf16_t*)Out)[o] = (bf16_t)val;
        } else {
          ((float*)Out)[o] = val + resid[o];
        }
      }
    }
  }
}

// ---------------------------------------------------------------------------
// Flash attention: one block per (b, h, 64-row q-block). 128 threads = 4 waves.
// Each wave owns 16 q rows; loop over 32-key blocks. K/V tiles are DOUBLE-
// BUFFERED and staged by the Tensor Data Mover: wave 0 issues the next
// block's DMA, then waits TENSORcnt<=2 so the current block is resident
// while the next two transfers stay in flight behind the 16 WMMAs.
// ---------------------------------------------------------------------------
#define AP 136      // K/V/Q LDS pitch (elems) == TDM row pitch with padding
#define PP 40       // P tile pitch

__global__ __launch_bounds__(128) void flash_attention_kernel(
    const bf16_t* __restrict__ Qh, const bf16_t* __restrict__ Kh,
    const bf16_t* __restrict__ Vh, bf16_t* __restrict__ Out) {
  __shared__ __align__(16) bf16_t Qlds[64][AP];
  __shared__ __align__(16) bf16_t Klds[2][32][AP];
  __shared__ __align__(16) bf16_t Vlds[2][32][AP];
  __shared__ __align__(16) bf16_t Plds[4][16][PP];

  int t = threadIdx.x, lane = t & 31, wave = t >> 5;
  int qb = blockIdx.x & 7;
  int h  = (blockIdx.x >> 3) & 7;
  int b  = blockIdx.x >> 6;

  const bf16_t* Qbase = Qh + ((size_t)b * N_LAT + qb * 64) * HD_DIM + h * DK_DIM;
  const bf16_t* Kbase = Kh + ((size_t)b * M_BYTE) * HD_DIM + h * DK_DIM;
  const bf16_t* Vbase = Vh + ((size_t)b * M_BYTE) * HD_DIM + h * DK_DIM;

  // LDS byte offsets of the TDM destination tiles (low 32 bits of the
  // generic pointer are the LDS aperture offset)
  unsigned koff[2] = {(unsigned)(unsigned long long)(void*)&Klds[0][0][0],
                      (unsigned)(unsigned long long)(void*)&Klds[1][0][0]};
  unsigned voff[2] = {(unsigned)(unsigned long long)(void*)&Vlds[0][0][0],
                      (unsigned)(unsigned long long)(void*)&Vlds[1][0][0]};

  // Prologue: wave 0 kicks off the first K/V block DMA into buffer 0
  if (wave == 0) {
    tdm_load_tile_2d(koff[0], Kbase, DK_DIM, M_BYTE, DK_DIM, 32, HD_DIM);
    tdm_load_tile_2d(voff[0], Vbase, DK_DIM, M_BYTE, DK_DIM, 32, HD_DIM);
  }

  // Stage the 64x128 Q tile once (1024 x uint4, 8 per thread)
#pragma unroll
  for (int i = 0; i < 8; ++i) {
    int idx = t + i * 128;
    int r = idx >> 4, c = (idx & 15) << 3;
    *(uint4*)(&Qlds[r][c]) = *(const uint4*)(Qbase + (size_t)r * HD_DIM + c);
  }

  v8f acc[8] = {};
  float mrow[8], lrow[8];
#pragma unroll
  for (int i = 0; i < 8; ++i) { mrow[i] = -1e30f; lrow[i] = 0.0f; }

  int qrow0 = wave * 16;
  int hf = lane >> 4, nn = lane & 15;

  for (int kb = 0; kb < M_BYTE; kb += 32) {
    int cur = (kb >> 5) & 1;
    if (wave == 0) {
      if (kb + 32 < M_BYTE) {
        // Issue next block into the other buffer (its readers finished at
        // the bottom barrier of iteration kb-32), then wait until only
        // those two newest transfers remain outstanding -> cur is resident.
        tdm_load_tile_2d(koff[cur ^ 1], Kbase + (size_t)(kb + 32) * HD_DIM,
                         DK_DIM, M_BYTE - kb - 32, DK_DIM, 32, HD_DIM);
        tdm_load_tile_2d(voff[cur ^ 1], Vbase + (size_t)(kb + 32) * HD_DIM,
                         DK_DIM, M_BYTE - kb - 32, DK_DIM, 32, HD_DIM);
        __builtin_amdgcn_s_wait_tensorcnt(2);
      } else {
        __builtin_amdgcn_s_wait_tensorcnt(0);
      }
    }
    __syncthreads();   // cur tiles visible to all 4 waves

    // S = Q * K^T for this wave's 16 rows x 32 keys (two 16x16 f32 frags)
    v8f s0 = {}, s1 = {};
#pragma unroll
    for (int d0 = 0; d0 < DK_DIM; d0 += 32) {
      v16bf aq  = load_a_frag(&Qlds[0][0], AP, qrow0, d0, lane);
      v16bf bk0 = load_bT_frag(&Klds[cur][0][0], AP, 0,  d0, lane);
      v16bf bk1 = load_bT_frag(&Klds[cur][0][0], AP, 16, d0, lane);
      s0 = wmma_bf16(aq, bk0, s0);
      s1 = wmma_bf16(aq, bk1, s1);
    }

    // Online softmax. Row r = i + 8*half lives in the 16 lanes of this half;
    // DPP16 butterflies reduce over columns without touching the LDS port.
#pragma unroll
    for (int i = 0; i < 8; ++i) {
      float x0 = s0[i] * TEMP_SC, x1 = s1[i] * TEMP_SC;
      float mx = half_reduce_max(fmaxf(x0, x1));
      float mnew = fmaxf(mrow[i], mx);
      float corr = exp2f((mrow[i] - mnew) * LOG2E);
      float p0 = exp2f((x0 - mnew) * LOG2E);
      float p1 = exp2f((x1 - mnew) * LOG2E);
      float rs = half_reduce_sum(p0 + p1);
      lrow[i] = lrow[i] * corr + rs;
      mrow[i] = mnew;
#pragma unroll
      for (int nv = 0; nv < 8; ++nv) acc[nv][i] *= corr;
      int prow = i + 8 * hf;
      Plds[wave][prow][nn]      = (bf16_t)p0;   // keys kb..kb+15
      Plds[wave][prow][16 + nn] = (bf16_t)p1;   // keys kb+16..kb+31
    }

    // acc += P(16x32) * V(32x128). P redistributed via wave-private LDS.
    v16bf ap = load_a_frag(&Plds[wave][0][0], PP, 0, 0, lane);
#pragma unroll
    for (int nv = 0; nv < 8; ++nv) {
      v16bf bv = load_b_frag(&Vlds[cur][0][0], AP, 0, nv * 16, lane);
      acc[nv] = wmma_bf16(ap, bv, acc[nv]);
    }
    __syncthreads();   // all waves done reading cur before it is re-DMA'd
  }

  // Normalize and store to attn buffer in [B*N_LAT, H*128] layout
  bf16_t* Obase = Out + ((size_t)b * N_LAT + qb * 64 + qrow0) * HD_DIM + h * DK_DIM;
#pragma unroll
  for (int nv = 0; nv < 8; ++nv) {
#pragma unroll
    for (int i = 0; i < 8; ++i) {
      int row = i + 8 * hf;
      float v = acc[nv][i] / lrow[i];
      Obase[(size_t)row * HD_DIM + nv * 16 + nn] = (bf16_t)v;
    }
  }
}

// ---------------------------------------------------------------------------
// Host-side orchestration
// ---------------------------------------------------------------------------
static inline void conv_launch(const float* in, bf16_t* out, size_t n, hipStream_t s) {
  f32_to_bf16_kernel<<<dim3((unsigned)(n / 1024)), dim3(256), 0, s>>>(in, out);
}

extern "C" void kernel_launch(void* const* d_in, const int* in_sizes, int n_in,
                              void* d_out, int out_size, void* d_ws, size_t ws_size,
                              hipStream_t stream) {
  const float* q   = (const float*)d_in[0];
  const float* k   = (const float*)d_in[1];
  const float* v   = (const float*)d_in[2];
  const float* w_q = (const float*)d_in[3];
  const float* w_k = (const float*)d_in[4];
  const float* w_v = (const float*)d_in[5];
  const float* w_o = (const float*)d_in[6];
  float* out = (float*)d_out;

  // Workspace carve-up (element counts)
  const size_t nQ  = (size_t)B_DIM * N_LAT * D_LATENT;      //  4M
  const size_t nKV = (size_t)B_DIM * M_BYTE * D_BYTE;       // 16M
  const size_t nWq = (size_t)D_LATENT * HD_DIM;             //  1M
  const size_t nWk = (size_t)D_BYTE * HD_DIM;               //  0.5M
  const size_t nH  = (size_t)B_DIM * N_LAT * HD_DIM;        //  4M
  const size_t nKH = (size_t)B_DIM * M_BYTE * HD_DIM;       // 32M

  bf16_t* p = (bf16_t*)d_ws;
  bf16_t* q_bf  = p; p += nQ;
  bf16_t* k_bf  = p; p += nKV;
  bf16_t* v_bf  = p; p += nKV;
  bf16_t* wq_bf = p; p += nWq;
  bf16_t* wk_bf = p; p += nWk;
  bf16_t* wv_bf = p; p += nWk;
  bf16_t* wo_bf = p; p += nWq;
  bf16_t* qh    = p; p += nH;
  bf16_t* kh    = p; p += nKH;
  bf16_t* vh    = p; p += nKH;
  bf16_t* attn  = p; p += nH;

  // 1) fp32 -> bf16
  conv_launch(q,   q_bf,  nQ,  stream);
  conv_launch(k,   k_bf,  nKV, stream);
  conv_launch(v,   v_bf,  nKV, stream);
  conv_launch(w_q, wq_bf, nWq, stream);
  conv_launch(w_k, wk_bf, nWk, stream);
  conv_launch(w_v, wv_bf, nWk, stream);
  conv_launch(w_o, wo_bf, nWq, stream);

  // 2) Projections (row-major GEMMs; batch folded into M)
  {
    dim3 blk(256);
    // qh = q_bf[4096,1024] @ wq[1024,1024]
    gemm_bf16_kernel<true><<<dim3(HD_DIM / 128, (B_DIM * N_LAT) / 128), blk, 0, stream>>>(
        q_bf, wq_bf, nullptr, qh, B_DIM * N_LAT, HD_DIM, D_LATENT);
    // kh = k_bf[32768,512] @ wk[512,1024]
    gemm_bf16_kernel<true><<<dim3(HD_DIM / 128, (B_DIM * M_BYTE) / 128), blk, 0, stream>>>(
        k_bf, wk_bf, nullptr, kh, B_DIM * M_BYTE, HD_DIM, D_BYTE);
    // vh = v_bf[32768,512] @ wv[512,1024]
    gemm_bf16_kernel<true><<<dim3(HD_DIM / 128, (B_DIM * M_BYTE) / 128), blk, 0, stream>>>(
        v_bf, wv_bf, nullptr, vh, B_DIM * M_BYTE, HD_DIM, D_BYTE);
  }

  // 3) Flash attention: B * H * (512/64) = 512 blocks, 128 threads
  flash_attention_kernel<<<dim3(B_DIM * H_DIM * (N_LAT / 64)), dim3(128), 0, stream>>>(
      qh, kh, vh, attn);

  // 4) Output projection + residual: out = attn @ wo + q
  gemm_bf16_kernel<false><<<dim3(D_LATENT / 128, (B_DIM * N_LAT) / 128), dim3(256), 0, stream>>>(
      attn, wo_bf, q, out, B_DIM * N_LAT, D_LATENT, HD_DIM);
}
